// LSTM_2834678415413
// MI455X (gfx1250) — matrix operand order repaired and checked by reading, not compiled, not run on hardware
//
#include <hip/hip_runtime.h>

// ---------------------------------------------------------------------------
// 2-layer LSTM (B=128, T=512, H=512, IN=OUT=1, FUT=16) for gfx1250 (MI455X).
// Strategy: bf16 WMMA (v_wmma_f32_16x16x32_bf16) for the three per-step
// [128x512]x[512x2048] GEMMs, fp32 accumulation + fp32 cell state.
// Weights converted to bf16 once per call; they live in L2 (6 MB << 192 MB).
// 528 steps x 3 kernels chained on the stream (graph-capture safe).
// ---------------------------------------------------------------------------

#define HID   512
#define BATCH 128
#define TSEQ  512
#define FUTS  16
#define TOTT  (TSEQ + FUTS)   // 528
#define G4    (4 * HID)       // 2048

typedef __bf16 bfv16 __attribute__((ext_vector_type(16)));
typedef __bf16 bfv8  __attribute__((ext_vector_type(8)));
typedef float  fv8   __attribute__((ext_vector_type(8)));

__device__ __forceinline__ unsigned short f2bf(float f) {
    unsigned int u = __float_as_uint(f);
    unsigned int r = u + 0x7FFFu + ((u >> 16) & 1u);   // round-to-nearest-even
    return (unsigned short)(r >> 16);
}
__device__ __forceinline__ float sigf(float x)  { return 1.0f / (1.0f + __expf(-x)); }
__device__ __forceinline__ float tanhf_(float x){ return 2.0f / (1.0f + __expf(-2.0f * x)) - 1.0f; }

// A fragment (16x32 bf16, M x K) from row-major [*, HID] bf16 activations.
// lane<16: rows m0+lane, K = k0+{0..7} and k0+{16..23}
// lane>=16: rows m0+lane-16, K = k0+{8..15} and k0+{24..31}
__device__ __forceinline__ bfv16 load_a(const unsigned short* __restrict__ base,
                                        int m0, int k0, int lane) {
    const int l15 = lane & 15;
    const int o   = (lane & 16) ? 8 : 0;
    const unsigned short* p = base + (m0 + l15) * HID + k0 + o;
    bfv16 a;
    ((bfv8*)&a)[0] = *(const bfv8*)(p);        // 16B
    ((bfv8*)&a)[1] = *(const bfv8*)(p + 16);   // 16B
    return a;
}

// B fragment (32x16 bf16, K x N) where B[k][n] = W[col0+n][k], W row-major [G4][HID].
// lane<16: col n=lane, K = k0+{0..7} and k0+{16..23};  lane>=16: col n=lane-16, K = k0+{8..15} and k0+{24..31}
__device__ __forceinline__ bfv16 load_b(const unsigned short* __restrict__ w,
                                        int col0, int k0, int lane) {
    const int l15 = lane & 15;
    const int ko  = (lane & 16) ? 8 : 0;
    const unsigned short* p = w + (col0 + l15) * HID + k0 + ko;
    bfv16 b;
    ((bfv8*)&b)[0] = *(const bfv8*)(p);
    ((bfv8*)&b)[1] = *(const bfv8*)(p + 16);
    return b;
}

#define WMMA_BF16(a, b, c) \
    __builtin_amdgcn_wmma_f32_16x16x32_bf16(false, (a), false, (b), (short)0, (c), false, false)

// ---------------------------------------------------------------------------
// Prep: fp32->bf16 weight conversion, fused biases, zero-init states.
// grid 4096 x 256 covers 2048*512 elements.
// ---------------------------------------------------------------------------
__global__ __launch_bounds__(256) void lstm_prep(
    const float* __restrict__ whh1, const float* __restrict__ wih2,
    const float* __restrict__ whh2,
    const float* __restrict__ bih1, const float* __restrict__ bhh1,
    const float* __restrict__ bih2, const float* __restrict__ bhh2,
    unsigned short* __restrict__ w1b,  unsigned short* __restrict__ wi2b,
    unsigned short* __restrict__ wh2b,
    float* __restrict__ bias1, float* __restrict__ bias2,
    float* __restrict__ c1, float* __restrict__ c2,
    unsigned short* __restrict__ h1b0, unsigned short* __restrict__ h1b1,
    unsigned short* __restrict__ h2b0, unsigned short* __restrict__ h2b1)
{
    const int idx = blockIdx.x * blockDim.x + threadIdx.x;
    w1b[idx]  = f2bf(whh1[idx]);
    wi2b[idx] = f2bf(wih2[idx]);
    wh2b[idx] = f2bf(whh2[idx]);
    if (idx < G4) {
        bias1[idx] = bih1[idx] + bhh1[idx];
        bias2[idx] = bih2[idx] + bhh2[idx];
    }
    if (idx < BATCH * HID) {
        c1[idx] = 0.f; c2[idx] = 0.f;
        h1b0[idx] = 0; h1b1[idx] = 0; h2b0[idx] = 0; h2b1[idx] = 0;
    }
}

// ---------------------------------------------------------------------------
// Layer 1: g = h1_prev @ w_hh1^T + x_t (x) w_ih1 + bias1 ; cell update.
// grid(32,4), block 64 (2 waves). Wave owns M-tile; block owns 16 hidden cols.
// ---------------------------------------------------------------------------
__global__ __launch_bounds__(64) void lstm_l1(
    const unsigned short* __restrict__ h1_prev,   // [B][H] bf16
    const unsigned short* __restrict__ w1,        // [4H][H] bf16
    const float* __restrict__ wih1,               // [4H]
    const float* __restrict__ bias1,              // [4H]
    const float* __restrict__ xsrc, int xpitch, int xoff,
    float* __restrict__ c1,                       // [B][H]
    unsigned short* __restrict__ h1_next)         // [B][H] bf16
{
    const int lane = threadIdx.x & 31;
    const int wave = threadIdx.x >> 5;
    const int m0 = (blockIdx.y * 2 + wave) * 16;
    const int n0 = blockIdx.x * 16;

    fv8 acc0 = {}, acc1 = {}, acc2 = {}, acc3 = {};
    #pragma unroll 4
    for (int k0 = 0; k0 < HID; k0 += 32) {
        bfv16 a  = load_a(h1_prev, m0, k0, lane);
        bfv16 b0 = load_b(w1, 0 * HID + n0, k0, lane);
        acc0 = WMMA_BF16(a, b0, acc0);
        bfv16 b1 = load_b(w1, 1 * HID + n0, k0, lane);
        acc1 = WMMA_BF16(a, b1, acc1);
        bfv16 b2 = load_b(w1, 2 * HID + n0, k0, lane);
        acc2 = WMMA_BF16(a, b2, acc2);
        bfv16 b3 = load_b(w1, 3 * HID + n0, k0, lane);
        acc3 = WMMA_BF16(a, b3, acc3);
    }

    const int n = n0 + (lane & 15);
    const float wi_i = wih1[n],           bi_i = bias1[n];
    const float wi_f = wih1[HID + n],     bi_f = bias1[HID + n];
    const float wi_g = wih1[2 * HID + n], bi_g = bias1[2 * HID + n];
    const float wi_o = wih1[3 * HID + n], bi_o = bias1[3 * HID + n];
    const int mbase = m0 + ((lane & 16) ? 8 : 0);
    #pragma unroll
    for (int v = 0; v < 8; ++v) {
        const int m = mbase + v;
        const float xv = xsrc[m * xpitch + xoff];
        const float gi = acc0[v] + xv * wi_i + bi_i;
        const float gf = acc1[v] + xv * wi_f + bi_f;
        const float gg = acc2[v] + xv * wi_g + bi_g;
        const float go = acc3[v] + xv * wi_o + bi_o;
        const int idx = m * HID + n;
        const float cn = sigf(gf) * c1[idx] + sigf(gi) * tanhf_(gg);
        c1[idx] = cn;
        h1_next[idx] = f2bf(sigf(go) * tanhf_(cn));
    }
}

// ---------------------------------------------------------------------------
// Layer 2: g = h1_new @ w_ih2^T + h2_prev @ w_hh2^T + bias2 ; cell update.
// Also stores h2 in fp32 for the output projection.
// ---------------------------------------------------------------------------
__global__ __launch_bounds__(64) void lstm_l2(
    const unsigned short* __restrict__ h1_new,    // [B][H] bf16
    const unsigned short* __restrict__ h2_prev,   // [B][H] bf16
    const unsigned short* __restrict__ wi2,       // [4H][H] bf16
    const unsigned short* __restrict__ wh2,       // [4H][H] bf16
    const float* __restrict__ bias2,              // [4H]
    float* __restrict__ c2,                       // [B][H]
    unsigned short* __restrict__ h2_next,         // [B][H] bf16
    float* __restrict__ h2f)                      // [B][H] fp32
{
    const int lane = threadIdx.x & 31;
    const int wave = threadIdx.x >> 5;
    const int m0 = (blockIdx.y * 2 + wave) * 16;
    const int n0 = blockIdx.x * 16;

    fv8 acc0 = {}, acc1 = {}, acc2 = {}, acc3 = {};
    #pragma unroll 2
    for (int k0 = 0; k0 < HID; k0 += 32) {
        bfv16 a1 = load_a(h1_new,  m0, k0, lane);
        bfv16 a2 = load_a(h2_prev, m0, k0, lane);
        bfv16 b;
        b = load_b(wi2, 0 * HID + n0, k0, lane);  acc0 = WMMA_BF16(a1, b, acc0);
        b = load_b(wh2, 0 * HID + n0, k0, lane);  acc0 = WMMA_BF16(a2, b, acc0);
        b = load_b(wi2, 1 * HID + n0, k0, lane);  acc1 = WMMA_BF16(a1, b, acc1);
        b = load_b(wh2, 1 * HID + n0, k0, lane);  acc1 = WMMA_BF16(a2, b, acc1);
        b = load_b(wi2, 2 * HID + n0, k0, lane);  acc2 = WMMA_BF16(a1, b, acc2);
        b = load_b(wh2, 2 * HID + n0, k0, lane);  acc2 = WMMA_BF16(a2, b, acc2);
        b = load_b(wi2, 3 * HID + n0, k0, lane);  acc3 = WMMA_BF16(a1, b, acc3);
        b = load_b(wh2, 3 * HID + n0, k0, lane);  acc3 = WMMA_BF16(a2, b, acc3);
    }

    const int n = n0 + (lane & 15);
    const float bi_i = bias2[n];
    const float bi_f = bias2[HID + n];
    const float bi_g = bias2[2 * HID + n];
    const float bi_o = bias2[3 * HID + n];
    const int mbase = m0 + ((lane & 16) ? 8 : 0);
    #pragma unroll
    for (int v = 0; v < 8; ++v) {
        const int m = mbase + v;
        const float gi = acc0[v] + bi_i;
        const float gf = acc1[v] + bi_f;
        const float gg = acc2[v] + bi_g;
        const float go = acc3[v] + bi_o;
        const int idx = m * HID + n;
        const float cn = sigf(gf) * c2[idx] + sigf(gi) * tanhf_(gg);
        c2[idx] = cn;
        const float hn = sigf(go) * tanhf_(cn);
        h2_next[idx] = f2bf(hn);
        h2f[idx] = hn;
    }
}

// ---------------------------------------------------------------------------
// Output projection: out[b, t] = dot(h2f[b,:], w_out) + b_out. One block.
// ---------------------------------------------------------------------------
__global__ __launch_bounds__(256) void lstm_out(
    const float* __restrict__ h2f, const float* __restrict__ wout,
    const float* __restrict__ bout, float* __restrict__ out, int t)
{
    __shared__ float red[256];
    const int tid  = threadIdx.x;
    const int b    = tid >> 1;
    const int half = tid & 1;
    const float* hr = h2f  + b * HID + half * 256;
    const float* wr = wout + half * 256;
    float s = 0.f;
    #pragma unroll 8
    for (int k = 0; k < 256; ++k) s += hr[k] * wr[k];
    red[tid] = s;
    __syncthreads();
    if (!half) out[b * TOTT + t] = red[tid] + red[tid + 1] + bout[0];
}

// ---------------------------------------------------------------------------
extern "C" void kernel_launch(void* const* d_in, const int* in_sizes, int n_in,
                              void* d_out, int out_size, void* d_ws, size_t ws_size,
                              hipStream_t stream)
{
    const float* x    = (const float*)d_in[0];   // [128][512]
    const float* wih1 = (const float*)d_in[1];   // [2048][1]
    const float* whh1 = (const float*)d_in[2];   // [2048][512]
    const float* bih1 = (const float*)d_in[3];   // [2048]
    const float* bhh1 = (const float*)d_in[4];
    const float* wih2 = (const float*)d_in[5];   // [2048][512]
    const float* whh2 = (const float*)d_in[6];   // [2048][512]
    const float* bih2 = (const float*)d_in[7];
    const float* bhh2 = (const float*)d_in[8];
    const float* wout = (const float*)d_in[9];   // [1][512]
    const float* bout = (const float*)d_in[10];  // [1]
    float* out = (float*)d_out;                  // [128][528]

    char* p = (char*)d_ws;
    auto carve = [&](size_t bytes) -> char* {
        char* r = p;
        p += (bytes + 255) & ~(size_t)255;
        return r;
    };
    unsigned short* w1b   = (unsigned short*)carve((size_t)G4 * HID * 2);
    unsigned short* wi2b  = (unsigned short*)carve((size_t)G4 * HID * 2);
    unsigned short* wh2b  = (unsigned short*)carve((size_t)G4 * HID * 2);
    float*          bias1 = (float*)carve(G4 * 4);
    float*          bias2 = (float*)carve(G4 * 4);
    float*          c1    = (float*)carve(BATCH * HID * 4);
    float*          c2    = (float*)carve(BATCH * HID * 4);
    float*          h2f   = (float*)carve(BATCH * HID * 4);
    unsigned short* h1b[2], *h2b[2];
    h1b[0] = (unsigned short*)carve(BATCH * HID * 2);
    h1b[1] = (unsigned short*)carve(BATCH * HID * 2);
    h2b[0] = (unsigned short*)carve(BATCH * HID * 2);
    h2b[1] = (unsigned short*)carve(BATCH * HID * 2);

    lstm_prep<<<(G4 * HID) / 256, 256, 0, stream>>>(
        whh1, wih2, whh2, bih1, bhh1, bih2, bhh2,
        w1b, wi2b, wh2b, bias1, bias2, c1, c2,
        h1b[0], h1b[1], h2b[0], h2b[1]);

    const dim3 grid(HID / 16, 4);   // 32 hidden-col chunks x 4 M-chunks
    const dim3 block(64);           // 2 waves, each owns one 16-row M-tile

    int cur = 0;
    for (int t = 0; t < TOTT; ++t) {
        const float* xsrc;
        int xpitch, xoff;
        if (t < TSEQ) { xsrc = x;   xpitch = TSEQ; xoff = t; }
        else          { xsrc = out; xpitch = TOTT; xoff = t - 1; }  // feed back prev output

        lstm_l1<<<grid, block, 0, stream>>>(
            h1b[cur], w1b, wih1, bias1, xsrc, xpitch, xoff, c1, h1b[cur ^ 1]);
        lstm_l2<<<grid, block, 0, stream>>>(
            h1b[cur ^ 1], h2b[cur], wi2b, wh2b, bias2, c2, h2b[cur ^ 1], h2f);
        lstm_out<<<1, 256, 0, stream>>>(h2f, wout, bout, out, t);
        cur ^= 1;
    }
}
